// LinearCrossAttention_89593017794890
// MI455X (gfx1250) — matrix operand order repaired and checked by reading, not compiled, not run on hardware
//
#include <hip/hip_runtime.h>
#include <hip/hip_bf16.h>

typedef __attribute__((ext_vector_type(16))) _Float16 v16h;
typedef __attribute__((ext_vector_type(8)))  _Float16 v8h;
typedef __attribute__((ext_vector_type(4)))  _Float16 v4h;
typedef __attribute__((ext_vector_type(8)))  float    v8f;
typedef __attribute__((ext_vector_type(4)))  unsigned v4u;
typedef __attribute__((ext_vector_type(8)))  int      v8i;
typedef __attribute__((ext_vector_type(4)))  int      v4i;

#define WMMA_F16(a, b, c) \
  __builtin_amdgcn_wmma_f32_16x16x32_f16(false, (a), false, (b), (short)0, (c), false, false)

#define CAT16(lo, hi) \
  __builtin_shufflevector((lo), (hi), 0, 1, 2, 3, 4, 5, 6, 7, 8, 9, 10, 11, 12, 13, 14, 15)

#if __has_builtin(__builtin_amdgcn_tensor_load_to_lds) && \
    __has_builtin(__builtin_amdgcn_s_wait_tensorcnt)
#define HAVE_TDM 1
#endif

// ---------------------------------------------------------------------------
// TDM 2D tile load: rows x 32 halves from a K-contiguous f16 tensor into LDS,
// with 8-half row padding (LDS row stride 40 halves). D# packed per ISA §8.3/8.4:
//  g0: [1:0]=count=1, [63:32]=lds_addr, [120:64]=global_addr, [127:126]=type 2
//  g1: [17:16]=data_size(2B), [20]=pad_enable, [24:22]=pad_interval(3 => 16 DW),
//      [31:25]=pad_amount(3 => 4 DW), [79:48]=tensor_dim0, [111:80]=tensor_dim1,
//      [127:112]=tile_dim0=32, [143:128]=tile_dim1=rows, [207:160]=dim0_stride
// ---------------------------------------------------------------------------
__device__ __forceinline__ void tdm_load_tile32(const void* gaddr, unsigned lds_off,
                                                unsigned row_len, unsigned rows) {
#if defined(HAVE_TDM)
  unsigned long long ga = (unsigned long long)gaddr;
  v4u g0 = {1u, lds_off, (unsigned)ga,
            (unsigned)((ga >> 32) & 0x1FFFFFFu) | (2u << 30)};
  v8i g1 = {(int)((1u << 16) | (1u << 20) | (3u << 22) | (3u << 25)),
            (int)((row_len & 0xFFFFu) << 16),
            (int)(((row_len >> 16) & 0xFFFFu) | ((rows & 0xFFFFu) << 16)),
            (int)(((rows >> 16) & 0xFFFFu) | (32u << 16)),
            (int)rows,
            (int)row_len,
            0, 0};
  v4i z4 = {0, 0, 0, 0};
#if __clang_major__ >= 23
  v8i z8 = {0, 0, 0, 0, 0, 0, 0, 0};
  __builtin_amdgcn_tensor_load_to_lds(g0, g1, z4, z4, z8, 0);
#else
  __builtin_amdgcn_tensor_load_to_lds(g0, g1, z4, z4, 0);
#endif
#endif
}

// ---------------------------------------------------------------------------
// Fragment loaders per CDNA5 ISA VGPR layouts (CDNA5 ISA §7.12.2).
// ---------------------------------------------------------------------------
__device__ __forceinline__ v16h ld_fragA(const _Float16* p0, int stride, int m0, int k0) {
  const int lane = threadIdx.x & 31;
  const _Float16* p = p0 + (size_t)(m0 + (lane & 15)) * stride + k0 + (lane >> 4) * 8;
  v8h lo = *(const v8h*)p;
  v8h hi = *(const v8h*)(p + 16);
  return CAT16(lo, hi);
}

__device__ __forceinline__ v16h ld_fragB(const _Float16* p0, int stride, int k0, int n0) {
  const int lane = threadIdx.x & 31;
  const _Float16* p = p0 + (size_t)(n0 + (lane & 15)) * stride + k0 + (lane >> 4) * 8;
  v8h lo = *(const v8h*)p;
  v8h hi = *(const v8h*)(p + 16);
  return CAT16(lo, hi);
}

// ===========================================================================
// Kernel 0: weight prep — fp32 [rows x cols] -> f16 transposed [cols x rows]
// ===========================================================================
__global__ __launch_bounds__(256)
void wprep_kernel(const float* __restrict__ in, _Float16* __restrict__ out,
                  int rows, int cols) {
  int idx = blockIdx.x * 256 + threadIdx.x;
  if (idx >= rows * cols) return;
  int r = idx / cols, c = idx % cols;
  out[(size_t)c * rows + r] = (_Float16)in[idx];
}

// ===========================================================================
// Kernel 1: q = softmax_head(LN(x @ Wq)) * SCALE   -> f16
// LDS layout: sO f32 @0 (65536) | sA @65536 (2560) | sB @68096 (40960)
// ===========================================================================
__global__ __launch_bounds__(256)
void qproj_kernel(const float* __restrict__ x, const _Float16* __restrict__ WqT,
                  const float* __restrict__ g, const float* __restrict__ bb,
                  _Float16* __restrict__ qout) {
  extern __shared__ char smem[];
  float*    sO = (float*)smem;
  _Float16* sA = (_Float16*)(smem + 32 * 512 * 4);
  _Float16* sB = sA + 32 * 40;
  const unsigned LDS_B_OFF = 32 * 512 * 4 + 32 * 40 * 2;

  const int t = threadIdx.x;
  const int wave = t >> 5, lane = t & 31;
  const int hi = lane >> 4, nn = lane & 15;
  const int row0 = blockIdx.x * 32;
  const int wm = wave >> 2, wn = wave & 3;

  const v8f zero8 = {0.f, 0.f, 0.f, 0.f, 0.f, 0.f, 0.f, 0.f};
  v8f acc[8];
#pragma unroll
  for (int i = 0; i < 8; ++i) acc[i] = zero8;

  for (int k0 = 0; k0 < 1024; k0 += 32) {
#if defined(HAVE_TDM)
    if (wave == 0) tdm_load_tile32(&WqT[k0], LDS_B_OFF, 1024u, 512u);
#else
#pragma unroll
    for (int s = t; s < 512 * 4; s += 256) {
      int n = s >> 2, part = (s & 3) * 8;
      *(v8h*)&sB[n * 40 + part] = *(const v8h*)&WqT[(size_t)n * 1024 + k0 + part];
    }
#endif
    { // stage A: 32x32 fp32 -> f16
      int r = t >> 3, c4 = (t & 7) * 4;
      float4 f = *(const float4*)&x[(size_t)(row0 + r) * 1024 + k0 + c4];
      v4h h = {(_Float16)f.x, (_Float16)f.y, (_Float16)f.z, (_Float16)f.w};
      *(v4h*)&sA[r * 40 + c4] = h;
    }
    if (k0 + 32 < 1024)
      __builtin_prefetch(&x[(size_t)(row0 + (t & 31)) * 1024 + k0 + 32], 0, 0);
#if defined(HAVE_TDM)
    if (wave == 0) __builtin_amdgcn_s_wait_tensorcnt(0);
#endif
    __syncthreads();
    v16h af = ld_fragA(sA, 40, wm * 16, 0);
#pragma unroll
    for (int nt = 0; nt < 8; ++nt) {
      v16h bf = ld_fragB(sB, 40, 0, wn * 128 + nt * 16);
      acc[nt] = WMMA_F16(af, bf, acc[nt]);
    }
    __syncthreads();
  }

#pragma unroll
  for (int nt = 0; nt < 8; ++nt)
#pragma unroll
    for (int i = 0; i < 8; ++i) {
      int r = wm * 16 + hi * 8 + i;
      sO[r * 512 + wn * 128 + nt * 16 + nn] = acc[nt][i];
    }
  __syncthreads();

  for (int rr = 0; rr < 4; ++rr) {
    int r = wave * 4 + rr;
    float v[16];
    float s = 0.f, s2 = 0.f;
#pragma unroll
    for (int i = 0; i < 16; ++i) {
      float xv = sO[r * 512 + lane * 16 + i];
      v[i] = xv; s += xv; s2 += xv * xv;
    }
    for (int m = 16; m >= 1; m >>= 1) { s += __shfl_xor(s, m, 32); s2 += __shfl_xor(s2, m, 32); }
    float mu = s * (1.f / 512.f);
    float rs = rsqrtf(s2 * (1.f / 512.f) - mu * mu + 1e-5f);
    float mx = -1e30f;
#pragma unroll
    for (int i = 0; i < 16; ++i) {
      int c = lane * 16 + i;
      v[i] = (v[i] - mu) * rs * g[c] + bb[c];
      mx = fmaxf(mx, v[i]);
    }
    mx = fmaxf(mx, __shfl_xor(mx, 1, 32));
    mx = fmaxf(mx, __shfl_xor(mx, 2, 32));
    float es = 0.f;
#pragma unroll
    for (int i = 0; i < 16; ++i) { v[i] = __expf(v[i] - mx); es += v[i]; }
    es += __shfl_xor(es, 1, 32);
    es += __shfl_xor(es, 2, 32);
    float inv = 8.f / es;
#pragma unroll
    for (int i = 0; i < 16; ++i)
      qout[(size_t)(row0 + r) * 512 + lane * 16 + i] = (_Float16)(v[i] * inv);
  }
}

// ===========================================================================
// Kernel 2: kv = context @ Wkv (fp32 out). LDS: sA @0 (5120) | sB @5120
// ===========================================================================
__global__ __launch_bounds__(256)
void kvproj_kernel(const float* __restrict__ ctx_in, const _Float16* __restrict__ WkvT,
                   float* __restrict__ kv) {
  extern __shared__ char smem[];
  _Float16* sA = (_Float16*)smem;
  _Float16* sB = sA + 64 * 40;
  const unsigned LDS_B_OFF = 64 * 40 * 2;

  const int t = threadIdx.x;
  const int wave = t >> 5, lane = t & 31;
  const int hi = lane >> 4, nn = lane & 15;
  const int row0 = blockIdx.x * 64, col0 = blockIdx.y * 128;
  const int wm = wave >> 2, wn = wave & 3;

  const v8f zero8 = {0.f, 0.f, 0.f, 0.f, 0.f, 0.f, 0.f, 0.f};
  v8f acc[2][2];
#pragma unroll
  for (int a = 0; a < 2; ++a)
#pragma unroll
    for (int b = 0; b < 2; ++b) acc[a][b] = zero8;

  for (int k0 = 0; k0 < 1024; k0 += 32) {
#if defined(HAVE_TDM)
    if (wave == 0)
      tdm_load_tile32(&WkvT[(size_t)col0 * 1024 + k0], LDS_B_OFF, 1024u, 128u);
#else
#pragma unroll
    for (int s = t; s < 128 * 4; s += 256) {
      int n = s >> 2, part = (s & 3) * 8;
      *(v8h*)&sB[n * 40 + part] = *(const v8h*)&WkvT[(size_t)(col0 + n) * 1024 + k0 + part];
    }
#endif
#pragma unroll
    for (int i = t; i < 64 * 8; i += 256) {
      int r = i >> 3, c4 = (i & 7) * 4;
      float4 f = *(const float4*)&ctx_in[(size_t)(row0 + r) * 1024 + k0 + c4];
      v4h h = {(_Float16)f.x, (_Float16)f.y, (_Float16)f.z, (_Float16)f.w};
      *(v4h*)&sA[r * 40 + c4] = h;
    }
    if (k0 + 32 < 1024)
      __builtin_prefetch(&ctx_in[(size_t)(row0 + (t & 63)) * 1024 + k0 + 32], 0, 0);
#if defined(HAVE_TDM)
    if (wave == 0) __builtin_amdgcn_s_wait_tensorcnt(0);
#endif
    __syncthreads();
    v16h af0 = ld_fragA(sA, 40, wm * 32, 0);
    v16h af1 = ld_fragA(sA, 40, wm * 32 + 16, 0);
#pragma unroll
    for (int nt = 0; nt < 2; ++nt) {
      v16h bf = ld_fragB(sB, 40, 0, wn * 32 + nt * 16);
      acc[0][nt] = WMMA_F16(af0, bf, acc[0][nt]);
      acc[1][nt] = WMMA_F16(af1, bf, acc[1][nt]);
    }
    __syncthreads();
  }

#pragma unroll
  for (int mt = 0; mt < 2; ++mt)
#pragma unroll
    for (int nt = 0; nt < 2; ++nt)
#pragma unroll
      for (int i = 0; i < 8; ++i) {
        int row = row0 + wm * 32 + mt * 16 + hi * 8 + i;
        int col = col0 + wn * 32 + nt * 16 + nn;
        kv[(size_t)row * 1024 + col] = acc[mt][nt][i];
      }
}

// ===========================================================================
// Kernel 3: colmax[b, c] = max_n k[b,n,c] (incl. null_k)
// ===========================================================================
__global__ __launch_bounds__(256)
void colmax_kernel(const float* __restrict__ kv, const float* __restrict__ null_kv,
                   float* __restrict__ colmax) {
  __shared__ float red[256];
  const int b = blockIdx.x >> 3, cg = blockIdx.x & 7;
  const int t = threadIdx.x;
  const int cl = t & 63, rr = t >> 6;
  const int c = cg * 64 + cl;
  float mx = null_kv[cl];
  const float* base = kv + (size_t)b * 4096 * 1024 + c;
  for (int n = rr; n < 4096; n += 4) mx = fmaxf(mx, base[(size_t)n * 1024]);
  red[t] = mx;
  __syncthreads();
  if (rr == 0)
    colmax[b * 512 + c] =
        fmaxf(fmaxf(red[cl], red[cl + 64]), fmaxf(red[cl + 128], red[cl + 192]));
}

// ===========================================================================
// Kernel 4: ctxT[b,h,e,d] = ((sum_n exp(k-max)*v + null) / denom)  (transposed)
// ===========================================================================
__global__ __launch_bounds__(256)
void ctx_kernel(const float* __restrict__ kv, const float* __restrict__ null_kv,
                const float* __restrict__ colmax, _Float16* __restrict__ ctxo) {
  __shared__ float kc[8 * 64];
  __shared__ float vc[8 * 64];
  const int bh = blockIdx.x, b = bh >> 3, h = bh & 7;
  const int t = threadIdx.x;
  const int d = t >> 2, e0 = (t & 3) * 16;
  const float cmax = colmax[b * 512 + h * 64 + d];
  float acc[16];
#pragma unroll
  for (int i = 0; i < 16; ++i) acc[i] = 0.f;
  float denom = 0.f;

  const size_t rowbase = (size_t)b * 4096 * 1024;
  for (int n0 = 0; n0 < 4096; n0 += 8) {
#pragma unroll
    for (int j = 0; j < 4; ++j) {
      int i = t + 256 * j;
      int half = i >> 9;
      int r = (i >> 6) & 7, cc = i & 63;
      float val = kv[rowbase + (size_t)(n0 + r) * 1024 + half * 512 + h * 64 + cc];
      if (half == 0) kc[r * 64 + cc] = val; else vc[r * 64 + cc] = val;
    }
    __syncthreads();
#pragma unroll
    for (int r = 0; r < 8; ++r) {
      float ek = __expf(kc[r * 64 + d] - cmax);
      denom += ek;
      const float* vr = &vc[r * 64 + e0];
#pragma unroll
      for (int i = 0; i < 16; ++i) acc[i] += ek * vr[i];
    }
    __syncthreads();
  }
  float ekn = __expf(null_kv[d] - cmax);
  denom += ekn;
#pragma unroll
  for (int i = 0; i < 16; ++i) acc[i] += ekn * null_kv[64 + e0 + i];
  float inv = 1.f / denom;
  _Float16* o = ctxo + (size_t)bh * 64 * 64;
#pragma unroll
  for (int i = 0; i < 16; ++i) o[(size_t)(e0 + i) * 64 + d] = (_Float16)(acc[i] * inv);
}

// ===========================================================================
// Kernel 5: out = LN(q @ ctx per head) @ Wout
// LDS: sT f32 @0 (65536) | sQ/sB @65536 (40960) | sTh @106496 (33280)
// ===========================================================================
__global__ __launch_bounds__(256)
void out_kernel(const _Float16* __restrict__ qws, const _Float16* __restrict__ ctxT,
                const float* __restrict__ g, const float* __restrict__ bb,
                const _Float16* __restrict__ WoutT, float* __restrict__ out) {
  extern __shared__ char smem[];
  float*    sT  = (float*)smem;
  _Float16* sQ  = (_Float16*)(smem + 32 * 512 * 4);
  _Float16* sB  = sQ;
  _Float16* sTh = (_Float16*)(smem + 32 * 512 * 4 + 512 * 40 * 2);
  const unsigned LDS_B_OFF = 32 * 512 * 4;

  const int t = threadIdx.x;
  const int wave = t >> 5, lane = t & 31;
  const int hi = lane >> 4, nn = lane & 15;
  const int row0 = blockIdx.x * 32;
  const int b = row0 >> 12;

  // ---- Phase 1: tmp = q @ ctx per head; wave = head ----
#pragma unroll
  for (int s = t; s < 32 * 64; s += 256) {
    int r = s >> 6, c = (s & 63) * 8;
    *(v8h*)&sQ[r * 520 + c] = *(const v8h*)&qws[(size_t)(row0 + r) * 512 + c];
  }
  __syncthreads();

  const int h = wave;
  const _Float16* ctxh = ctxT + (size_t)(b * 8 + h) * 64 * 64;
  const v8f zero8 = {0.f, 0.f, 0.f, 0.f, 0.f, 0.f, 0.f, 0.f};
  v8f acc1[2][4];
#pragma unroll
  for (int a = 0; a < 2; ++a)
#pragma unroll
    for (int c = 0; c < 4; ++c) acc1[a][c] = zero8;

#pragma unroll
  for (int ks = 0; ks < 2; ++ks) {
    v16h af0 = ld_fragA(sQ, 520, 0,  h * 64 + ks * 32);
    v16h af1 = ld_fragA(sQ, 520, 16, h * 64 + ks * 32);
#pragma unroll
    for (int nt = 0; nt < 4; ++nt) {
      v16h bf = ld_fragB(ctxh, 64, ks * 32, nt * 16);  // global, L2-hot
      acc1[0][nt] = WMMA_F16(af0, bf, acc1[0][nt]);
      acc1[1][nt] = WMMA_F16(af1, bf, acc1[1][nt]);
    }
  }
#pragma unroll
  for (int mt = 0; mt < 2; ++mt)
#pragma unroll
    for (int nt = 0; nt < 4; ++nt)
#pragma unroll
      for (int i = 0; i < 8; ++i) {
        int r = mt * 16 + hi * 8 + i;
        sT[r * 512 + h * 64 + nt * 16 + nn] = acc1[mt][nt][i];
      }
  __syncthreads();

  // ---- LayerNorm rows -> sTh (f16) ----
  for (int rr = 0; rr < 4; ++rr) {
    int r = wave * 4 + rr;
    float v[16];
    float s = 0.f, s2 = 0.f;
#pragma unroll
    for (int i = 0; i < 16; ++i) {
      float xv = sT[r * 512 + lane * 16 + i];
      v[i] = xv; s += xv; s2 += xv * xv;
    }
    for (int m = 16; m >= 1; m >>= 1) { s += __shfl_xor(s, m, 32); s2 += __shfl_xor(s2, m, 32); }
    float mu = s * (1.f / 512.f);
    float rs = rsqrtf(s2 * (1.f / 512.f) - mu * mu + 1e-5f);
#pragma unroll
    for (int i = 0; i < 16; ++i) {
      int c = lane * 16 + i;
      sTh[r * 520 + c] = (_Float16)((v[i] - mu) * rs * g[c] + bb[c]);
    }
  }
  __syncthreads();

  // ---- Phase 2: out = sTh @ Wout, two 512-wide N halves ----
  const int wm = wave >> 2, wn = wave & 3;
  for (int nh = 0; nh < 2; ++nh) {
    v8f acc2[8];
#pragma unroll
    for (int i = 0; i < 8; ++i) acc2[i] = zero8;
    for (int k0 = 0; k0 < 512; k0 += 32) {
#if defined(HAVE_TDM)
      if (wave == 0)
        tdm_load_tile32(&WoutT[(size_t)nh * 512 * 512 + k0], LDS_B_OFF, 512u, 512u);
      if (wave == 0) __builtin_amdgcn_s_wait_tensorcnt(0);
#else
#pragma unroll
      for (int s = t; s < 512 * 4; s += 256) {
        int n = s >> 2, part = (s & 3) * 8;
        *(v8h*)&sB[n * 40 + part] =
            *(const v8h*)&WoutT[(size_t)(nh * 512 + n) * 512 + k0 + part];
      }
#endif
      __syncthreads();
      v16h af = ld_fragA(sTh, 520, wm * 16, k0);
#pragma unroll
      for (int nt = 0; nt < 8; ++nt) {
        v16h bf = ld_fragB(sB, 40, 0, wn * 128 + nt * 16);
        acc2[nt] = WMMA_F16(af, bf, acc2[nt]);
      }
      __syncthreads();
    }
#pragma unroll
    for (int nt = 0; nt < 8; ++nt)
#pragma unroll
      for (int i = 0; i < 8; ++i) {
        int row = row0 + wm * 16 + hi * 8 + i;
        int col = nh * 512 + wn * 128 + nt * 16 + nn;
        out[(size_t)row * 1024 + col] = acc2[nt][i];
      }
  }
}

// ===========================================================================
extern "C" void kernel_launch(void* const* d_in, const int* in_sizes, int n_in,
                              void* d_out, int out_size, void* d_ws, size_t ws_size,
                              hipStream_t stream) {
  const float* x       = (const float*)d_in[0];
  const float* context = (const float*)d_in[1];
  const float* Wq      = (const float*)d_in[2];
  const float* Wkv     = (const float*)d_in[3];
  const float* null_kv = (const float*)d_in[4];
  const float* ln_q_g  = (const float*)d_in[5];
  const float* ln_q_b  = (const float*)d_in[6];
  const float* ln_o_g  = (const float*)d_in[7];
  const float* ln_o_b  = (const float*)d_in[8];
  const float* Wout    = (const float*)d_in[9];
  float* outp = (float*)d_out;

  char* ws = (char*)d_ws;
  size_t off = 0;
  float*    kvbuf  = (float*)(ws + off);    off += (size_t)16384 * 1024 * 4;
  _Float16* qbuf   = (_Float16*)(ws + off); off += (size_t)16384 * 512 * 2;
  float*    cmax   = (float*)(ws + off);    off += (size_t)4 * 512 * 4;
  _Float16* ctxbuf = (_Float16*)(ws + off); off += (size_t)32 * 64 * 64 * 2;
  _Float16* WqT    = (_Float16*)(ws + off); off += (size_t)1024 * 512 * 2;
  _Float16* WkvT   = (_Float16*)(ws + off); off += (size_t)1024 * 1024 * 2;
  _Float16* WoutT  = (_Float16*)(ws + off); off += (size_t)512 * 1024 * 2;
  (void)ws_size; (void)in_sizes; (void)n_in; (void)out_size;

  const dim3 blk(256);
  const size_t smem1 = 32 * 512 * 4 + (32 * 40 + 512 * 40) * 2;     // ~109 KB
  const size_t smem2 = (64 * 40 + 128 * 40) * 2;                    // ~15 KB
  const size_t smem5 = 32 * 512 * 4 + 512 * 40 * 2 + 32 * 520 * 2;  // ~140 KB

  wprep_kernel<<<(1024 * 512 + 255) / 256, blk, 0, stream>>>(Wq, WqT, 1024, 512);
  wprep_kernel<<<(1024 * 1024 + 255) / 256, blk, 0, stream>>>(Wkv, WkvT, 1024, 1024);
  wprep_kernel<<<(512 * 1024 + 255) / 256, blk, 0, stream>>>(Wout, WoutT, 512, 1024);

  qproj_kernel<<<512, blk, smem1, stream>>>(x, WqT, ln_q_g, ln_q_b, qbuf);
  kvproj_kernel<<<dim3(256, 8), blk, smem2, stream>>>(context, WkvT, kvbuf);
  colmax_kernel<<<32, blk, 0, stream>>>(kvbuf, null_kv, cmax);
  ctx_kernel<<<32, blk, 0, stream>>>(kvbuf, null_kv, cmax, ctxbuf);
  out_kernel<<<512, blk, smem5, stream>>>(qbuf, ctxbuf, ln_o_g, ln_o_b, WoutT, outp);
}
